// EmbedRRN_54855322305083
// MI455X (gfx1250) — compile-verified, matrix operand order checked by reading, not run on hardware
//
#include <hip/hip_runtime.h>
#include <hip/hip_bf16.h>

typedef __attribute__((ext_vector_type(16))) _Float16 v16h;
typedef __attribute__((ext_vector_type(8)))  float    v8f;

#define SROWS 41472          // B*N = 512*81
#define HD 96
#define TILES 2592           // SROWS/16
#define ITERS 4

// ---------------------------------------------------------------------------
// helpers
// ---------------------------------------------------------------------------

// intra-wave LDS store->load ordering (CDNA5 split counters)
__device__ __forceinline__ void wave_lds_sync() {
  __builtin_amdgcn_wave_barrier();
  asm volatile("s_wait_dscnt 0" ::: "memory");
  __builtin_amdgcn_wave_barrier();
}

// Load a 16x32 f16 WMMA fragment (A-operand layout; B uses the same mapping
// against a pre-transposed weight matrix WT[n][k]).
// p points at row 0 of the 16-row tile, ld = row stride in elements.
__device__ __forceinline__ v16h load_frag(const _Float16* p, int ld, int k0) {
  const int lane = threadIdx.x & 31;
  const int half = lane >> 4;
  const int r    = lane & 15;
  const _Float16* base = p + r * ld + k0 + half * 8;
  v16h f;
#pragma unroll
  for (int e = 0; e < 8; ++e) f[e]     = base[e];        // K = k0 + half*8 + e
#pragma unroll
  for (int e = 0; e < 8; ++e) f[8 + e] = base[16 + e];   // K = k0 + 16 + half*8 + e
  return f;
}

__device__ __forceinline__ v8f wmma_f16(v16h a, v16h b, v8f c) {
  return __builtin_amdgcn_wmma_f32_16x16x32_f16(false, a, false, b, (short)0, c,
                                                false, false);
}

// 16x16 output tile of (16xK) x (KxN) with K=96; A row-major (lda), BT = W^T rows n.
__device__ __forceinline__ v8f tile_gemm96(const _Float16* A, int lda,
                                           const _Float16* BT, int ldb, v8f acc) {
#pragma unroll
  for (int k0 = 0; k0 < 96; k0 += 32) {
    v16h a = load_frag(A, lda, k0);
    v16h b = load_frag(BT, ldb, k0);
    acc = wmma_f16(a, b, acc);
  }
  return acc;
}

__device__ __forceinline__ v8f bias_acc(const float* b, int n0) {
  float v = b[n0 + (threadIdx.x & 15)];
  v8f a;
#pragma unroll
  for (int r = 0; r < 8; ++r) a[r] = v;
  return a;
}

__device__ __forceinline__ v8f zero_acc() {
  v8f a;
#pragma unroll
  for (int r = 0; r < 8; ++r) a[r] = 0.f;
  return a;
}

// store 16x16 f32 D-tile to row-major buffer
__device__ __forceinline__ void store_tile_f32(float* p, int ld, v8f d) {
  const int half = (threadIdx.x >> 4) & 1;
  const int n    = threadIdx.x & 15;
#pragma unroll
  for (int r = 0; r < 8; ++r) p[(r + 8 * half) * ld + n] = d[r];
}

__device__ __forceinline__ void store_tile_f16(_Float16* p, int ld, v8f d, bool relu) {
  const int half = (threadIdx.x >> 4) & 1;
  const int n    = threadIdx.x & 15;
#pragma unroll
  for (int r = 0; r < 8; ++r) {
    float v = d[r];
    if (relu) v = v > 0.f ? v : 0.f;
    p[(r + 8 * half) * ld + n] = (_Float16)v;
  }
}

// ---------------------------------------------------------------------------
// weight prep: WT[n*Kp + k] = (k<K) ? W[k*N + n] : 0   (f32 -> f16, transpose)
// ---------------------------------------------------------------------------
__global__ void transpose_w_kernel(const float* __restrict__ W,
                                   _Float16* __restrict__ WT,
                                   int K, int N, int Kp) {
  int total = N * Kp;
  for (int idx = blockIdx.x * blockDim.x + threadIdx.x; idx < total;
       idx += gridDim.x * blockDim.x) {
    int n = idx / Kp, k = idx - n * Kp;
    WT[idx] = (_Float16)((k < K) ? W[(size_t)k * N + n] : 0.f);
  }
}

__global__ void copy_c_kernel(const float* __restrict__ src, float* __restrict__ dst,
                              int n) {
  for (int i = blockIdx.x * blockDim.x + threadIdx.x; i < n;
       i += gridDim.x * blockDim.x)
    dst[i] = src[i];
}

// ---------------------------------------------------------------------------
// embed + input MLP:  X = mlp3(embed_w[grids]);  h = X  (one 16-row tile/wave)
// ---------------------------------------------------------------------------
__global__ __launch_bounds__(32) void embed_kernel(
    const int* __restrict__ grids, const float* __restrict__ embed_w,
    const _Float16* __restrict__ WT_in1 /*96x32 zero-padded*/,
    const _Float16* __restrict__ WT_in2, const _Float16* __restrict__ WT_in3,
    const float* __restrict__ in_b1, const float* __restrict__ in_b2,
    const float* __restrict__ in_b3,
    _Float16* __restrict__ X16, _Float16* __restrict__ h16) {
  __shared__ _Float16 sE[16 * 32];
  __shared__ _Float16 sA[16 * 96];
  __shared__ _Float16 sB[16 * 96];
  const int lane = threadIdx.x;
  const int row0 = blockIdx.x * 16;

  {  // stage embeddings, zero-padded K 16->32
    int r = lane & 15, ch = lane >> 4;
    int g = grids[row0 + r];
    _Float16* srow = sE + r * 32;
#pragma unroll
    for (int cc = 0; cc < 16; ++cc) {
      int c = ch * 16 + cc;
      srow[c] = (_Float16)((c < 16) ? embed_w[g * 16 + c] : 0.f);
    }
  }
  wave_lds_sync();
#pragma unroll
  for (int nt = 0; nt < 6; ++nt) {  // layer 1 (K=32 incl. zero pad)
    v8f acc = bias_acc(in_b1, nt * 16);
    acc = wmma_f16(load_frag(sE, 32, 0), load_frag(WT_in1 + nt * 16 * 32, 32, 0), acc);
    store_tile_f16(sA + nt * 16, 96, acc, true);
  }
  wave_lds_sync();
#pragma unroll
  for (int nt = 0; nt < 6; ++nt) {  // layer 2
    v8f acc = bias_acc(in_b2, nt * 16);
    acc = tile_gemm96(sA, 96, WT_in2 + nt * 16 * 96, 96, acc);
    store_tile_f16(sB + nt * 16, 96, acc, true);
  }
  wave_lds_sync();
#pragma unroll
  for (int nt = 0; nt < 6; ++nt) {  // layer 3 (no relu) -> X, h0
    v8f acc = bias_acc(in_b3, nt * 16);
    acc = tile_gemm96(sB, 96, WT_in3 + nt * 16 * 96, 96, acc);
    store_tile_f16(X16 + (size_t)row0 * 96 + nt * 16, 96, acc, false);
    store_tile_f16(h16 + (size_t)row0 * 96 + nt * 16, 96, acc, false);
  }
}

// ---------------------------------------------------------------------------
// XG = X @ g_w1a + g_b1   (precomputed once; X is loop-invariant)
// ---------------------------------------------------------------------------
__global__ __launch_bounds__(32) void xg_kernel(const _Float16* __restrict__ X16,
                                                const _Float16* __restrict__ WT_g1a,
                                                const float* __restrict__ g_b1,
                                                float* __restrict__ XG) {
  const int row0 = blockIdx.x * 16;
  const _Float16* xrow = X16 + (size_t)row0 * 96;
#pragma unroll
  for (int nt = 0; nt < 6; ++nt) {
    v8f acc = bias_acc(g_b1, nt * 16);
    acc = tile_gemm96(xrow, 96, WT_g1a + nt * 16 * 96, 96, acc);
    store_tile_f32(XG + (size_t)row0 * 96 + nt * 16, 96, acc);
  }
}

// ---------------------------------------------------------------------------
// A = h @ f_w1a + f_b1 ;  Bm = h @ f_w1b   (edge-MLP layer-1 separability)
// ---------------------------------------------------------------------------
__global__ __launch_bounds__(32) void ab_kernel(const _Float16* __restrict__ h16,
                                                const _Float16* __restrict__ WT_f1a,
                                                const _Float16* __restrict__ WT_f1b,
                                                const float* __restrict__ f_b1,
                                                _Float16* __restrict__ Ah,
                                                _Float16* __restrict__ Bm) {
  const int row0 = blockIdx.x * 16;
  const _Float16* hrow = h16 + (size_t)row0 * 96;
#pragma unroll
  for (int nt = 0; nt < 6; ++nt) {
    v8f a = bias_acc(f_b1, nt * 16);
    a = tile_gemm96(hrow, 96, WT_f1a + nt * 16 * 96, 96, a);
    store_tile_f16(Ah + (size_t)row0 * 96 + nt * 16, 96, a, false);
    v8f b = zero_acc();
    b = tile_gemm96(hrow, 96, WT_f1b + nt * 16 * 96, 96, b);
    store_tile_f16(Bm + (size_t)row0 * 96 + nt * 16, 96, b, false);
  }
}

// ---------------------------------------------------------------------------
// edge kernel: 4 nodes/block (80 edges = 5 exact row-tiles), 4 waves.
// t1 = relu(A_i + Bm_j); t2 = relu(t1@f_w2+b2); t3 = t2@f_w3+b3; M_i = sum_j t3
// ---------------------------------------------------------------------------
__global__ __launch_bounds__(128) void edge_kernel(
    const _Float16* __restrict__ Ah, const _Float16* __restrict__ Bm,
    const int* __restrict__ nbr,
    const _Float16* __restrict__ WT_f2, const _Float16* __restrict__ WT_f3,
    const float* __restrict__ f_b2, const float* __restrict__ f_b3,
    _Float16* __restrict__ Mout) {
  __shared__ _Float16 stage1[4][16 * 96];
  __shared__ _Float16 stage2[4][16 * 96];
  __shared__ float Msum[4 * 96];

  const int tid = threadIdx.x;
  const int lane = tid & 31;
  const int wave = tid >> 5;
  const int node0 = blockIdx.x * 4;

  for (int idx = tid; idx < 4 * 96; idx += 128) Msum[idx] = 0.f;
  __syncthreads();

  for (int t = wave; t < 5; t += 4) {  // 5 tiles over 4 waves (wave0 does 2)
    {  // build stage1: pre-act = A[src] + Bm[dst] (A carries f_b1), relu
      int r = lane & 15, ch = lane >> 4;
      int e = 16 * t + r;
      int q = e / 20, d = e - q * 20;
      int g = node0 + q;
      int b = g / 81, i = g - b * 81;
      int j = b * 81 + nbr[i * 20 + d];
      const _Float16* arow = Ah + (size_t)g * 96;
      const _Float16* brow = Bm + (size_t)j * 96;
      _Float16* srow = &stage1[wave][r * 96];
#pragma unroll
      for (int cc = 0; cc < 48; ++cc) {
        int c = ch * 48 + cc;
        float v = (float)arow[c] + (float)brow[c];
        srow[c] = (_Float16)(v > 0.f ? v : 0.f);
      }
    }
    wave_lds_sync();
    v8f acc[6];  // layer 2
#pragma unroll
    for (int nt = 0; nt < 6; ++nt) {
      acc[nt] = bias_acc(f_b2, nt * 16);
      acc[nt] = tile_gemm96(&stage1[wave][0], 96, WT_f2 + nt * 16 * 96, 96, acc[nt]);
    }
#pragma unroll
    for (int nt = 0; nt < 6; ++nt)
      store_tile_f16(&stage2[wave][0] + nt * 16, 96, acc[nt], true);
    wave_lds_sync();
#pragma unroll
    for (int nt = 0; nt < 6; ++nt) {  // layer 3 + segmented neighbor sum
      v8f o = bias_acc(f_b3, nt * 16);
      o = tile_gemm96(&stage2[wave][0], 96, WT_f3 + nt * 16 * 96, 96, o);
      const int half = lane >> 4, n = lane & 15;
#pragma unroll
      for (int r = 0; r < 8; ++r) {
        int e = 16 * t + r + 8 * half;
        int q = e / 20;
        atomicAdd(&Msum[q * 96 + nt * 16 + n], o[r]);
      }
    }
  }
  __syncthreads();
  for (int idx = tid; idx < 4 * 96; idx += 128)
    Mout[(size_t)node0 * 96 + idx] = (_Float16)Msum[idx];
}

// ---------------------------------------------------------------------------
// node kernel: g-MLP, LSTM gates + update, readout MLP. One wave / 16 rows.
// ---------------------------------------------------------------------------
__global__ __launch_bounds__(32) void node_kernel(
    const _Float16* __restrict__ M16, const float* __restrict__ XG,
    const _Float16* __restrict__ WT_g1b, const _Float16* __restrict__ WT_g2,
    const _Float16* __restrict__ WT_g3, const float* __restrict__ g_b2,
    const float* __restrict__ g_b3, const _Float16* __restrict__ WT_wih,
    const _Float16* __restrict__ WT_whh, const float* __restrict__ bih,
    const float* __restrict__ bhh, _Float16* __restrict__ h16,
    float* __restrict__ c32, const _Float16* __restrict__ WT_r1,
    const _Float16* __restrict__ WT_r2, const _Float16* __restrict__ WT_r3,
    const float* __restrict__ r_b1, const float* __restrict__ r_b2,
    const float* __restrict__ r_b3, float* __restrict__ out_it) {
  __shared__ _Float16 sA[16 * 96];
  __shared__ _Float16 sB[16 * 96];
  __shared__ _Float16 sH[16 * 96];
  __shared__ float gbuf[16 * 384];

  const int lane = threadIdx.x;
  const int half = (lane >> 4) & 1;
  const int n = lane & 15;
  const int row0 = blockIdx.x * 16;

  // u1 = relu(XG + M @ g_w1b)   (XG carries X@g_w1a + g_b1)
#pragma unroll
  for (int nt = 0; nt < 6; ++nt) {
    v8f acc;
#pragma unroll
    for (int r = 0; r < 8; ++r)
      acc[r] = XG[(size_t)(row0 + r + 8 * half) * 96 + nt * 16 + n];
    acc = tile_gemm96(M16 + (size_t)row0 * 96, 96, WT_g1b + nt * 16 * 96, 96, acc);
    store_tile_f16(sA + nt * 16, 96, acc, true);
  }
  wave_lds_sync();
#pragma unroll
  for (int nt = 0; nt < 6; ++nt) {  // u2
    v8f acc = bias_acc(g_b2, nt * 16);
    acc = tile_gemm96(sA, 96, WT_g2 + nt * 16 * 96, 96, acc);
    store_tile_f16(sB + nt * 16, 96, acc, true);
  }
  wave_lds_sync();
#pragma unroll
  for (int nt = 0; nt < 6; ++nt) {  // x_in (no relu) -> sA
    v8f acc = bias_acc(g_b3, nt * 16);
    acc = tile_gemm96(sB, 96, WT_g3 + nt * 16 * 96, 96, acc);
    store_tile_f16(sA + nt * 16, 96, acc, false);
  }
  wave_lds_sync();

  // gates = x_in @ wih + h @ whh + (bih + bhh)  (24 N-tiles of 16)
  const _Float16* hrow = h16 + (size_t)row0 * 96;
  for (int nt = 0; nt < 24; ++nt) {
    v8f acc;
    float bsum = bih[nt * 16 + n] + bhh[nt * 16 + n];
#pragma unroll
    for (int r = 0; r < 8; ++r) acc[r] = bsum;
    acc = tile_gemm96(sA, 96, WT_wih + nt * 16 * 96, 96, acc);
    acc = tile_gemm96(hrow, 96, WT_whh + nt * 16 * 96, 96, acc);
    store_tile_f32(gbuf + nt * 16, 384, acc);
  }
  wave_lds_sync();

  // LSTM elementwise
  for (int idx = lane; idx < 16 * 96; idx += 32) {
    int m = idx / 96, col = idx - m * 96;
    float ig = gbuf[m * 384 + col];
    float fg = gbuf[m * 384 + 96 + col];
    float gg = gbuf[m * 384 + 192 + col];
    float og = gbuf[m * 384 + 288 + col];
    size_t gi = (size_t)(row0 + m) * 96 + col;
    float cold = c32[gi];
    float si = 1.f / (1.f + __expf(-ig));
    float sf = 1.f / (1.f + __expf(-fg));
    float so = 1.f / (1.f + __expf(-og));
    float c2 = sf * cold + si * tanhf(gg);
    float h2 = so * tanhf(c2);
    c32[gi] = c2;
    h16[gi] = (_Float16)h2;
    sH[m * 96 + col] = (_Float16)h2;
  }
  wave_lds_sync();

  // readout MLP
#pragma unroll
  for (int nt = 0; nt < 6; ++nt) {
    v8f acc = bias_acc(r_b1, nt * 16);
    acc = tile_gemm96(sH, 96, WT_r1 + nt * 16 * 96, 96, acc);
    store_tile_f16(sA + nt * 16, 96, acc, true);
  }
  wave_lds_sync();
#pragma unroll
  for (int nt = 0; nt < 6; ++nt) {
    v8f acc = bias_acc(r_b2, nt * 16);
    acc = tile_gemm96(sA, 96, WT_r2 + nt * 16 * 96, 96, acc);
    store_tile_f16(sB + nt * 16, 96, acc, true);
  }
  wave_lds_sync();
  {
    v8f acc = bias_acc(r_b3, 0);
    acc = tile_gemm96(sB, 96, WT_r3, 96, acc);
#pragma unroll
    for (int r = 0; r < 8; ++r)
      out_it[(size_t)(row0 + r + 8 * half) * 16 + n] = acc[r];
  }
}

// ---------------------------------------------------------------------------
// host
// ---------------------------------------------------------------------------
static inline size_t algn(size_t x) { return (x + 255) & ~(size_t)255; }

extern "C" void kernel_launch(void* const* d_in, const int* in_sizes, int n_in,
                              void* d_out, int out_size, void* d_ws, size_t ws_size,
                              hipStream_t stream) {
  (void)in_sizes; (void)n_in; (void)out_size; (void)ws_size;

  const int*   grids  = (const int*)d_in[0];
  const int*   nbr    = (const int*)d_in[1];
  const float* c0     = (const float*)d_in[3];
  const float* emb    = (const float*)d_in[4];
  const float* in_w1  = (const float*)d_in[5];
  const float* in_b1  = (const float*)d_in[6];
  const float* in_w2  = (const float*)d_in[7];
  const float* in_b2  = (const float*)d_in[8];
  const float* in_w3  = (const float*)d_in[9];
  const float* in_b3  = (const float*)d_in[10];
  const float* f_w1   = (const float*)d_in[11];
  const float* f_b1   = (const float*)d_in[12];
  const float* f_w2   = (const float*)d_in[13];
  const float* f_b2   = (const float*)d_in[14];
  const float* f_w3   = (const float*)d_in[15];
  const float* f_b3   = (const float*)d_in[16];
  const float* g_w1   = (const float*)d_in[17];
  const float* g_b1   = (const float*)d_in[18];
  const float* g_w2   = (const float*)d_in[19];
  const float* g_b2   = (const float*)d_in[20];
  const float* g_w3   = (const float*)d_in[21];
  const float* g_b3   = (const float*)d_in[22];
  const float* wih    = (const float*)d_in[23];
  const float* whh    = (const float*)d_in[24];
  const float* bih    = (const float*)d_in[25];
  const float* bhh    = (const float*)d_in[26];
  const float* r_w1   = (const float*)d_in[27];
  const float* r_b1   = (const float*)d_in[28];
  const float* r_w2   = (const float*)d_in[29];
  const float* r_b2   = (const float*)d_in[30];
  const float* r_w3   = (const float*)d_in[31];
  const float* r_b3   = (const float*)d_in[32];
  float* out = (float*)d_out;

  char* ws = (char*)d_ws;
  size_t off = 0;
  auto take = [&](size_t bytes) { char* p = ws + off; off = algn(off + bytes); return p; };

  _Float16* WT_in1 = (_Float16*)take(96 * 32 * 2);
  _Float16* WT_in2 = (_Float16*)take(96 * 96 * 2);
  _Float16* WT_in3 = (_Float16*)take(96 * 96 * 2);
  _Float16* WT_f1a = (_Float16*)take(96 * 96 * 2);
  _Float16* WT_f1b = (_Float16*)take(96 * 96 * 2);
  _Float16* WT_f2  = (_Float16*)take(96 * 96 * 2);
  _Float16* WT_f3  = (_Float16*)take(96 * 96 * 2);
  _Float16* WT_g1a = (_Float16*)take(96 * 96 * 2);
  _Float16* WT_g1b = (_Float16*)take(96 * 96 * 2);
  _Float16* WT_g2  = (_Float16*)take(96 * 96 * 2);
  _Float16* WT_g3  = (_Float16*)take(96 * 96 * 2);
  _Float16* WT_wih = (_Float16*)take(384 * 96 * 2);
  _Float16* WT_whh = (_Float16*)take(384 * 96 * 2);
  _Float16* WT_r1  = (_Float16*)take(96 * 96 * 2);
  _Float16* WT_r2  = (_Float16*)take(96 * 96 * 2);
  _Float16* WT_r3  = (_Float16*)take(16 * 96 * 2);

  const size_t act16 = (size_t)SROWS * 96 * 2;
  const size_t act32 = (size_t)SROWS * 96 * 4;
  _Float16* X16 = (_Float16*)take(act16);
  _Float16* h16 = (_Float16*)take(act16);
  _Float16* Ah  = (_Float16*)take(act16);
  _Float16* Bmb = (_Float16*)take(act16);
  _Float16* M16 = (_Float16*)take(act16);
  float*    XG  = (float*)take(act32);
  float*    c32 = (float*)take(act32);

  auto tp = [&](const float* W, _Float16* WT, int K, int N, int Kp) {
    transpose_w_kernel<<<dim3(64), 256, 0, stream>>>(W, WT, K, N, Kp);
  };
  tp(in_w1, WT_in1, 16, 96, 32);
  tp(in_w2, WT_in2, 96, 96, 96);
  tp(in_w3, WT_in3, 96, 96, 96);
  tp(f_w1,            WT_f1a, 96, 96, 96);
  tp(f_w1 + 96 * 96,  WT_f1b, 96, 96, 96);
  tp(f_w2, WT_f2, 96, 96, 96);
  tp(f_w3, WT_f3, 96, 96, 96);
  tp(g_w1,            WT_g1a, 96, 96, 96);
  tp(g_w1 + 96 * 96,  WT_g1b, 96, 96, 96);
  tp(g_w2, WT_g2, 96, 96, 96);
  tp(g_w3, WT_g3, 96, 96, 96);
  tp(wih, WT_wih, 96, 384, 96);
  tp(whh, WT_whh, 96, 384, 96);
  tp(r_w1, WT_r1, 96, 96, 96);
  tp(r_w2, WT_r2, 96, 96, 96);
  tp(r_w3, WT_r3, 96, 16, 96);

  embed_kernel<<<TILES, 32, 0, stream>>>(grids, emb, WT_in1, WT_in2, WT_in3,
                                         in_b1, in_b2, in_b3, X16, h16);
  copy_c_kernel<<<512, 256, 0, stream>>>(c0, c32, SROWS * 96);
  xg_kernel<<<TILES, 32, 0, stream>>>(X16, WT_g1a, g_b1, XG);

  for (int it = 0; it < ITERS; ++it) {
    ab_kernel<<<TILES, 32, 0, stream>>>(h16, WT_f1a, WT_f1b, f_b1, Ah, Bmb);
    edge_kernel<<<SROWS / 4, 128, 0, stream>>>(Ah, Bmb, nbr, WT_f2, WT_f3,
                                               f_b2, f_b3, M16);
    node_kernel<<<TILES, 32, 0, stream>>>(M16, XG, WT_g1b, WT_g2, WT_g3, g_b2,
                                          g_b3, WT_wih, WT_whh, bih, bhh, h16,
                                          c32, WT_r1, WT_r2, WT_r3, r_b1, r_b2,
                                          r_b3, out + (size_t)it * SROWS * 16);
  }
}
